// Transformer_65996467470702
// MI455X (gfx1250) — matrix operand order repaired and checked by reading, not compiled, not run on hardware
//
#include <hip/hip_runtime.h>
#include <hip/hip_bf16.h>
#include <math.h>

// ---------------------------------------------------------------------------
// MI455X (gfx1250) transformer forward. All GEMMs via v_wmma_f32_16x16x32_bf16
// (wave32). fp32 residual stream; bf16 GEMM operands; fp32 accumulate.
// GEMM invariants (guaranteed by call sites / workspace padding):
//   M % 128 == 0, K % 32 == 0, N % 8 == 0, lda % 4 == 0, ldb % 8 == 0
// -> loader needs no scalar fallbacks at all.
// ---------------------------------------------------------------------------

typedef __bf16 bf16_t;
typedef __attribute__((ext_vector_type(16))) __bf16 v16bf;
typedef __attribute__((ext_vector_type(8)))  __bf16 v8bf;
typedef __attribute__((ext_vector_type(4)))  __bf16 v4bf;
typedef __attribute__((ext_vector_type(8)))  float  v8f;

#define BB 16
#define SS 256
#define DD 768
#define NH 8
#define HDIM 96
#define FF 3072
#define VV 32000
#define PSZ 14
#define RR (BB*SS)          // 4096 rows (B*S)
#define BHB (BB*NH)         // 128 batched heads
#define PATCH_K (3*PSZ*PSZ) // 588
#define PATCH_KP 608        // 588 padded up to multiple of 32 (19*32)

#define BM 128
#define BN 64
#define BK 32

__device__ __forceinline__ bf16_t f2bf(float f) {
  union { float f; unsigned u; } in; in.f = f;
  unsigned u = in.u;
  u += 0x7FFFu + ((u >> 16) & 1u);          // round-to-nearest-even
  union { unsigned short s; __bf16 b; } out; out.s = (unsigned short)(u >> 16);
  return out.b;
}

__device__ __forceinline__ float pe_val(int s, int d) {
  const float kc = -9.210340371976184f / (float)DD;  // -ln(10000)/D
  float e = expf((float)(d & ~1) * kc);
  float a = (float)s * e;
  return (d & 1) ? cosf(a) : sinf(a);
}

// ---------------------------------------------------------------------------
// Generic batched bf16 GEMM: C = act(alpha * A*B + bias) + resid
// A: [M,K] (lda), B: [K,N] (ldb) or [N,K] if transB, C: [M,N] (ldc).
// Batch z -> (z/bdiv, z%bdiv) with per-operand offsets so attention tensors
// laid out [B,S,H,HD] address per (b,h).
// 256 threads = 8 waves (4x2); each wave owns a 32x32 tile of C
// => 4x v_wmma_f32_16x16x32_bf16 per 32-wide K-step.
// ---------------------------------------------------------------------------
__global__ __launch_bounds__(256)
void gemm_bf16_kernel(const bf16_t* __restrict__ A, const bf16_t* __restrict__ Bm,
                      const float* __restrict__ bias, const float* __restrict__ resid,
                      float* __restrict__ Cf, bf16_t* __restrict__ Cb,
                      int M, int N, int K, int lda, int ldb, int ldc,
                      long oA1, long oA2, long oB1, long oB2, long oC1, long oC2,
                      int bdiv, int transB, int act, float alpha)
{
  __shared__ __attribute__((aligned(16))) __bf16 As[BM * BK];   // As[m][k], 8KB
  __shared__ __attribute__((aligned(16))) __bf16 Bs[BN * BK];   // Bs[n][k], 4KB

  const int z  = blockIdx.z;
  const int z1 = z / bdiv, z2 = z % bdiv;
  const bf16_t* Ab = A  + (long)z1 * oA1 + (long)z2 * oA2;
  const bf16_t* Bb = Bm + (long)z1 * oB1 + (long)z2 * oB2;
  const long    coff = (long)z1 * oC1 + (long)z2 * oC2;

  const int m0 = blockIdx.y * BM;
  const int n0 = blockIdx.x * BN;
  const int tid  = threadIdx.x;
  const int lane = tid & 31;
  const int wave = tid >> 5;
  const int wm = (wave >> 1) * 32;   // 0,32,64,96
  const int wn = (wave & 1) * 32;    // 0,32

  v8f acc00 = {}, acc01 = {}, acc10 = {}, acc11 = {};

  // cooperative load mappings
  const int arow = tid >> 1,  acol = (tid & 1) * 16;  // A: 16 elems/thread
  const int b_r0 = tid >> 3,  b_c0 = (tid & 7) * 8;   // B (K-major source)
  const int b_r1 = tid >> 2,  b_c1 = (tid & 3) * 8;   // B (N-major, transB)

  const bf16_t* aPtr = Ab + (long)(m0 + arow) * lda + acol;   // advances by BK
  for (int k0 = 0; k0 < K; k0 += BK) {
    { // A -> As[m][k]  (M%BM==0, K%BK==0 -> always fully in-bounds)
      const v4bf* src = (const v4bf*)(aPtr + k0);
      union { v4bf q[4]; v8bf h[2]; } u;
      u.q[0] = src[0]; u.q[1] = src[1]; u.q[2] = src[2]; u.q[3] = src[3];
      *(v8bf*)&As[arow * BK + acol]     = u.h[0];
      *(v8bf*)&As[arow * BK + acol + 8] = u.h[1];
    }
    if (!transB) { // B[k][n] -> Bs[n][k] (transpose in LDS)
      // group of 8 n-values never straddles N (N%8==0); k always in-bounds
      v8bf v;
      if (n0 + b_c0 < N) {
        v = *(const v8bf*)(Bb + (long)(k0 + b_r0) * ldb + n0 + b_c0);
      } else {
        v = v8bf{};   // zero tile outside N (e.g. PV GEMM, N=96)
      }
      #pragma unroll
      for (int j = 0; j < 8; ++j) Bs[(b_c0 + j) * BK + b_r0] = v[j];
    } else {       // B[n][k] -> Bs[n][k]
      v8bf v;
      if (n0 + b_r1 < N) {
        v = *(const v8bf*)(Bb + (long)(n0 + b_r1) * ldb + k0 + b_c1);
      } else {
        v = v8bf{};
      }
      *(v8bf*)&Bs[b_r1 * BK + b_c1] = v;
    }
    // prefetch next K-step tiles into cache while this step computes
    if (k0 + BK < K) {
      __builtin_prefetch((const void*)(aPtr + k0 + BK), 0, 1);
      if (!transB) {
        if (n0 + b_c0 < N)
          __builtin_prefetch((const void*)(Bb + (long)(k0 + BK + b_r0) * ldb + n0 + b_c0), 0, 1);
      } else {
        if (n0 + b_r1 < N)
          __builtin_prefetch((const void*)(Bb + (long)(n0 + b_r1) * ldb + k0 + BK + b_c1), 0, 1);
      }
    }
    __syncthreads();

    // Fragments per ISA 7.12.2 (wave32, 16-bit):
    // A 16x32: lanes 0-15 row M=lane, K 0-7 then 16-23; lanes 16-31: +8.
    // B 32x16: lanes 0-15 col N=lane, K 0-7 then 16-23; lanes 16-31: +8.
    const int r16 = lane & 15;
    const int hi  = lane >> 4;
    union Frag { v16bf v; v8bf h[2]; };
    Frag fa0, fa1, fb0, fb1;
    const int ka = hi * 8;
    fa0.h[0] = *(const v8bf*)&As[(wm + r16) * BK + ka];
    fa0.h[1] = *(const v8bf*)&As[(wm + r16) * BK + ka + 16];
    fa1.h[0] = *(const v8bf*)&As[(wm + 16 + r16) * BK + ka];
    fa1.h[1] = *(const v8bf*)&As[(wm + 16 + r16) * BK + ka + 16];
    const int kb = hi * 8;
    fb0.h[0] = *(const v8bf*)&Bs[(wn + r16) * BK + kb];
    fb0.h[1] = *(const v8bf*)&Bs[(wn + r16) * BK + kb + 16];
    fb1.h[0] = *(const v8bf*)&Bs[(wn + 16 + r16) * BK + kb];
    fb1.h[1] = *(const v8bf*)&Bs[(wn + 16 + r16) * BK + kb + 16];

    acc00 = __builtin_amdgcn_wmma_f32_16x16x32_bf16(false, fa0.v, false, fb0.v,
                                                    (short)0, acc00, false, false);
    acc01 = __builtin_amdgcn_wmma_f32_16x16x32_bf16(false, fa0.v, false, fb1.v,
                                                    (short)0, acc01, false, false);
    acc10 = __builtin_amdgcn_wmma_f32_16x16x32_bf16(false, fa1.v, false, fb0.v,
                                                    (short)0, acc10, false, false);
    acc11 = __builtin_amdgcn_wmma_f32_16x16x32_bf16(false, fa1.v, false, fb1.v,
                                                    (short)0, acc11, false, false);
    __syncthreads();
  }

  // C/D layout: vgpr r -> row M = r + 8*(lane>=16); col N = lane&15
  const int rb = (lane >> 4) * 8;
  const int cn = lane & 15;
  #pragma unroll
  for (int i = 0; i < 2; ++i) {
    #pragma unroll
    for (int jt = 0; jt < 2; ++jt) {
      v8f acc = i ? (jt ? acc11 : acc10) : (jt ? acc01 : acc00);
      const int gn = n0 + wn + jt * 16 + cn;
      if (gn < N) {                       // only PV GEMM (N=96) ever fails
        const float bv = bias ? bias[gn] : 0.f;
        #pragma unroll
        for (int r = 0; r < 8; ++r) {
          const int gm = m0 + wm + i * 16 + rb + r;   // always < M
          float v = acc[r] * alpha + bv;
          if (act == 1) v = 0.5f * v * (1.0f + erff(v * 0.70710678118654752f));
          const long idx = coff + (long)gm * ldc + gn;
          if (resid) v += resid[idx];
          if (Cf) Cf[idx] = v;
          if (Cb) Cb[idx] = f2bf(v);
        }
      }
    }
  }
}

// ---------------------------------------------------------------------------
// Elementwise / reduction kernels
// ---------------------------------------------------------------------------
__global__ void f32_to_bf16_kernel(const float* __restrict__ in,
                                   bf16_t* __restrict__ out, long n) {
  long i = (long)blockIdx.x * blockDim.x + threadIdx.x;
  if (i < n) out[i] = f2bf(in[i]);
}

// convert [srcRows, cols] fp32 -> [dstRows, cols] bf16, zero rows >= srcRows
__global__ void conv_pad_rows_kernel(const float* __restrict__ in,
                                     bf16_t* __restrict__ out,
                                     int srcRows, int dstRows, int cols) {
  long i = (long)blockIdx.x * blockDim.x + threadIdx.x;
  if (i >= (long)dstRows * cols) return;
  int row = (int)(i / cols);
  out[i] = (row < srcRows) ? f2bf(in[i]) : f2bf(0.f);
}

// writes x0 padded to PATCH_KP columns; pad columns = 0
__global__ void patchify_kernel(const float* __restrict__ img,
                                bf16_t* __restrict__ x0) {
  long idx = (long)blockIdx.x * blockDim.x + threadIdx.x;
  if (idx >= (long)RR * PATCH_KP) return;
  int j = (int)(idx % PATCH_KP);
  if (j >= PATCH_K) { x0[idx] = f2bf(0.f); return; }
  long rs = idx / PATCH_KP;
  int s = (int)(rs % SS), n = (int)(rs / SS);
  int c = j / (PSZ * PSZ), rem = j % (PSZ * PSZ);
  int pi = rem / PSZ, pj = rem % PSZ;
  int gy = s / 16, gx = s % 16;
  int row = gy * PSZ + pi, col = gx * PSZ + pj;
  float v = img[(((long)n * 3 + c) * 224 + row) * 224 + col];
  x0[idx] = f2bf(v);
}

__global__ void add_pe_kernel(float* __restrict__ x) {
  long idx = (long)blockIdx.x * blockDim.x + threadIdx.x;
  if (idx >= (long)RR * DD) return;
  int d = (int)(idx % DD);
  int s = (int)((idx / DD) % SS);
  x[idx] += pe_val(s, d);
}

__global__ void embed_kernel(const int* __restrict__ tgt,
                             const float* __restrict__ emb,
                             float* __restrict__ y) {
  long idx = (long)blockIdx.x * blockDim.x + threadIdx.x;
  if (idx >= (long)RR * DD) return;
  int d = (int)(idx % DD);
  long r = idx / DD;
  int tok = tgt[r];
  y[idx] = emb[(long)tok * DD + d] * 27.712812921102035f   // sqrt(768)
           + pe_val((int)(r % SS), d);
}

__global__ __launch_bounds__(256)
void ln_kernel(const float* __restrict__ x, const float* __restrict__ g,
               const float* __restrict__ b, bf16_t* __restrict__ out, int Dn) {
  const long row = blockIdx.x;
  const float* xr = x + row * Dn;
  const int tid = threadIdx.x;
  float s = 0.f, s2 = 0.f;
  for (int d = tid; d < Dn; d += 256) { float v = xr[d]; s += v; s2 += v * v; }
  __shared__ float r1[256], r2[256];
  r1[tid] = s; r2[tid] = s2; __syncthreads();
  for (int st = 128; st > 0; st >>= 1) {
    if (tid < st) { r1[tid] += r1[tid + st]; r2[tid] += r2[tid + st]; }
    __syncthreads();
  }
  const float mean = r1[0] / Dn;
  const float var  = r2[0] / Dn - mean * mean;
  const float inv  = rsqrtf(var + 1e-5f);
  for (int d = tid; d < Dn; d += 256)
    out[row * Dn + d] = f2bf((xr[d] - mean) * inv * g[d] + b[d]);
}

// scores: [BH, Sq, Sk] fp32 -> probs bf16 (masked entries -> 0)
__global__ __launch_bounds__(256)
void softmax_kernel(const float* __restrict__ scores, bf16_t* __restrict__ probs,
                    int Sq, int Sk, int causal) {
  const int q = blockIdx.x, bh = blockIdx.y;
  const long base = ((long)bh * Sq + q) * Sk;
  const int tid = threadIdx.x;           // Sk == 256 == blockDim
  const bool ok = !causal || (tid <= q);
  float v = ok ? scores[base + tid] : -1e30f;
  __shared__ float r[256];
  r[tid] = v; __syncthreads();
  for (int st = 128; st > 0; st >>= 1) {
    if (tid < st) r[tid] = fmaxf(r[tid], r[tid + st]);
    __syncthreads();
  }
  const float mx = r[0]; __syncthreads();
  const float e = ok ? expf(v - mx) : 0.f;
  r[tid] = e; __syncthreads();
  for (int st = 128; st > 0; st >>= 1) {
    if (tid < st) r[tid] += r[tid + st];
    __syncthreads();
  }
  probs[base + tid] = f2bf(e / r[0]);
}

__global__ __launch_bounds__(256)
void logsoftmax_kernel(float* __restrict__ x, int Vn) {
  float* xr = x + (long)blockIdx.x * Vn;
  const int tid = threadIdx.x;
  float mx = -1e30f;
  for (int i = tid; i < Vn; i += 256) mx = fmaxf(mx, xr[i]);
  __shared__ float r[256];
  r[tid] = mx; __syncthreads();
  for (int st = 128; st > 0; st >>= 1) {
    if (tid < st) r[tid] = fmaxf(r[tid], r[tid + st]);
    __syncthreads();
  }
  mx = r[0]; __syncthreads();
  float s = 0.f;
  for (int i = tid; i < Vn; i += 256) s += expf(xr[i] - mx);
  r[tid] = s; __syncthreads();
  for (int st = 128; st > 0; st >>= 1) {
    if (tid < st) r[tid] += r[tid + st];
    __syncthreads();
  }
  const float lse = mx + logf(r[0]);
  for (int i = tid; i < Vn; i += 256) xr[i] -= lse;
}

// ---------------------------------------------------------------------------
// Host orchestration
// ---------------------------------------------------------------------------
extern "C" void kernel_launch(void* const* d_in, const int* in_sizes, int n_in,
                              void* d_out, int out_size, void* d_ws, size_t ws_size,
                              hipStream_t stream) {
  (void)in_sizes; (void)n_in; (void)out_size; (void)ws_size;

  const float* img     = (const float*)d_in[0];
  const int*   tgt     = (const int*)  d_in[1];
  // d_in[2]=src_mask (all ones), d_in[3]=tgt_mask (causal) -> handled by flag
  const float* patch_w = (const float*)d_in[4];
  const float* patch_b = (const float*)d_in[5];
  const float* emb     = (const float*)d_in[6];
  const float* encP[2][10];
  for (int b = 0; b < 2; ++b)
    for (int j = 0; j < 10; ++j) encP[b][j] = (const float*)d_in[7 + b * 10 + j];
  const float* enc_lng = (const float*)d_in[27];
  const float* enc_lnb = (const float*)d_in[28];
  const float* decP[2][16];
  for (int b = 0; b < 2; ++b)
    for (int j = 0; j < 16; ++j) decP[b][j] = (const float*)d_in[29 + b * 16 + j];
  const float* dec_lng = (const float*)d_in[61];
  const float* dec_lnb = (const float*)d_in[62];
  const float* proj_w  = (const float*)d_in[63];
  const float* proj_b  = (const float*)d_in[64];

  char* wsp = (char*)d_ws;
  auto alloc = [&](size_t bytes) -> void* {
    void* r = (void*)wsp;
    wsp += (bytes + 255) & ~(size_t)255;
    return r;
  };
  auto conv = [&](const float* src, bf16_t* dst, long n) {
    f32_to_bf16_kernel<<<dim3((unsigned)((n + 255) / 256)), dim3(256), 0, stream>>>(src, dst, n);
  };

  // ---- bf16 weight staging ----
  // patch_w padded 588 -> 608 rows (zeros) so patch GEMM has K % 32 == 0
  bf16_t* pw_bf = (bf16_t*)alloc((size_t)PATCH_KP * DD * 2);
  conv_pad_rows_kernel<<<dim3((unsigned)(((long)PATCH_KP * DD + 255) / 256)), 256, 0, stream>>>(
      patch_w, pw_bf, PATCH_K, PATCH_KP, DD);

  static const int  eIdx[6] = {2, 3, 4, 5, 6, 8};
  static const long eSz[6]  = {(long)DD*DD,(long)DD*DD,(long)DD*DD,(long)DD*DD,(long)DD*FF,(long)FF*DD};
  bf16_t* eW[2][6];
  for (int b = 0; b < 2; ++b)
    for (int j = 0; j < 6; ++j) {
      eW[b][j] = (bf16_t*)alloc((size_t)eSz[j] * 2);
      conv(encP[b][eIdx[j]], eW[b][j], eSz[j]);
    }
  static const int  dIdx[10] = {2, 3, 4, 5, 6, 7, 8, 9, 12, 14};
  static const long dSz[10]  = {(long)DD*DD,(long)DD*DD,(long)DD*DD,(long)DD*DD,
                                (long)DD*DD,(long)DD*DD,(long)DD*DD,(long)DD*DD,
                                (long)DD*FF,(long)FF*DD};
  bf16_t* dW[2][10];
  for (int b = 0; b < 2; ++b)
    for (int j = 0; j < 10; ++j) {
      dW[b][j] = (bf16_t*)alloc((size_t)dSz[j] * 2);
      conv(decP[b][dIdx[j]], dW[b][j], dSz[j]);
    }
  bf16_t* projw_bf = (bf16_t*)alloc((size_t)DD * VV * 2);
  conv(proj_w, projw_bf, (long)DD * VV);

  // ---- activations ----
  bf16_t* x0_bf  = (bf16_t*)alloc((size_t)RR * PATCH_KP * 2);
  float*  xf     = (float*) alloc((size_t)RR * DD * 4);
  float*  yf     = (float*) alloc((size_t)RR * DD * 4);
  bf16_t* nrm_bf = (bf16_t*)alloc((size_t)RR * DD * 2);
  bf16_t* x_bf   = (bf16_t*)alloc((size_t)RR * DD * 2);
  bf16_t* q_bf   = (bf16_t*)alloc((size_t)RR * DD * 2);
  bf16_t* k_bf   = (bf16_t*)alloc((size_t)RR * DD * 2);
  bf16_t* v_bf   = (bf16_t*)alloc((size_t)RR * DD * 2);
  bf16_t* o_bf   = (bf16_t*)alloc((size_t)RR * DD * 2);
  bf16_t* hid_bf = (bf16_t*)alloc((size_t)RR * FF * 2);
  float*  scores = (float*) alloc((size_t)BHB * SS * SS * 4);
  bf16_t* probs  = (bf16_t*)alloc((size_t)BHB * SS * SS * 2);
  bf16_t* enc_bf = (bf16_t*)alloc((size_t)RR * DD * 2);

  auto gemm = [&](const bf16_t* A, const bf16_t* Bm, const float* bias,
                  const float* resid, float* Cf, bf16_t* Cb,
                  int M, int N, int K, int lda, int ldb, int ldc,
                  int nbatch, int bdiv,
                  long oA1, long oA2, long oB1, long oB2, long oC1, long oC2,
                  int transB, int act, float alpha) {
    dim3 grid((unsigned)((N + BN - 1) / BN), (unsigned)((M + BM - 1) / BM), (unsigned)nbatch);
    gemm_bf16_kernel<<<grid, 256, 0, stream>>>(A, Bm, bias, resid, Cf, Cb,
        M, N, K, lda, ldb, ldc, oA1, oA2, oB1, oB2, oC1, oC2,
        bdiv, transB, act, alpha);
  };

  const float inv_sqrt_hd = 0.10206207261596575f; // 1/sqrt(96)

  auto attention = [&](const bf16_t* qin, const bf16_t* kvin,
                       bf16_t* wq, bf16_t* wk, bf16_t* wv, bf16_t* wo,
                       float* stream_f, bf16_t* stream_b, int causal) {
    gemm(qin,  wq, nullptr, nullptr, nullptr, q_bf, RR, DD, DD, DD, DD, DD,
         1, 1, 0,0,0,0,0,0, 0, 0, 1.f);
    gemm(kvin, wk, nullptr, nullptr, nullptr, k_bf, RR, DD, DD, DD, DD, DD,
         1, 1, 0,0,0,0,0,0, 0, 0, 1.f);
    gemm(kvin, wv, nullptr, nullptr, nullptr, v_bf, RR, DD, DD, DD, DD, DD,
         1, 1, 0,0,0,0,0,0, 0, 0, 1.f);
    // scores[bh] = Q_bh (256x96) * K_bh^T -> [BH,256,256]
    gemm(q_bf, k_bf, nullptr, nullptr, scores, nullptr,
         SS, SS, HDIM, DD, DD, SS, BHB, NH,
         (long)SS * DD, HDIM, (long)SS * DD, HDIM,
         (long)NH * SS * SS, (long)SS * SS,
         1, 0, inv_sqrt_hd);
    softmax_kernel<<<dim3(SS, BHB), 256, 0, stream>>>(scores, probs, SS, SS, causal);
    // O_bh = P_bh (256x256) * V_bh (256x96) -> [B,S,H,HD]
    gemm(probs, v_bf, nullptr, nullptr, nullptr, o_bf,
         SS, HDIM, SS, SS, DD, DD, BHB, NH,
         (long)NH * SS * SS, (long)SS * SS, (long)SS * DD, HDIM,
         (long)SS * DD, HDIM,
         0, 0, 1.f);
    gemm(o_bf, wo, nullptr, stream_f, stream_f, stream_b, RR, DD, DD, DD, DD, DD,
         1, 1, 0,0,0,0,0,0, 0, 0, 1.f);
  };

  // ================= encoder =================
  patchify_kernel<<<dim3((unsigned)(((long)RR * PATCH_KP + 255) / 256)), 256, 0, stream>>>(img, x0_bf);
  gemm(x0_bf, pw_bf, patch_b, nullptr, xf, nullptr, RR, DD, PATCH_KP,
       PATCH_KP, DD, DD, 1, 1, 0,0,0,0,0,0, 0, 0, 1.f);
  add_pe_kernel<<<dim3((unsigned)(((long)RR * DD + 255) / 256)), 256, 0, stream>>>(xf);

  for (int b = 0; b < 2; ++b) {
    ln_kernel<<<RR, 256, 0, stream>>>(xf, encP[b][0], encP[b][1], nrm_bf, DD);
    attention(nrm_bf, nrm_bf, eW[b][0], eW[b][1], eW[b][2], eW[b][3], xf, x_bf, 0);
    // faithful to reference: FFN applied to x (not a second norm)
    gemm(x_bf, eW[b][4], encP[b][7], nullptr, nullptr, hid_bf, RR, FF, DD,
         DD, FF, FF, 1, 1, 0,0,0,0,0,0, 0, 1 /*gelu*/, 1.f);
    gemm(hid_bf, eW[b][5], encP[b][9], xf, xf, nullptr, RR, DD, FF,
         FF, DD, DD, 1, 1, 0,0,0,0,0,0, 0, 0, 1.f);
  }
  ln_kernel<<<RR, 256, 0, stream>>>(xf, enc_lng, enc_lnb, enc_bf, DD);

  // ================= decoder =================
  embed_kernel<<<dim3((unsigned)(((long)RR * DD + 255) / 256)), 256, 0, stream>>>(tgt, emb, yf);

  for (int b = 0; b < 2; ++b) {
    ln_kernel<<<RR, 256, 0, stream>>>(yf, decP[b][0], decP[b][1], nrm_bf, DD);
    attention(nrm_bf, nrm_bf, dW[b][0], dW[b][1], dW[b][2], dW[b][3], yf, nullptr, 1);
    attention(nrm_bf, enc_bf, dW[b][4], dW[b][5], dW[b][6], dW[b][7], yf, nullptr, 0);
    ln_kernel<<<RR, 256, 0, stream>>>(yf, decP[b][10], decP[b][11], nrm_bf, DD);
    gemm(nrm_bf, dW[b][8], decP[b][13], nullptr, nullptr, hid_bf, RR, FF, DD,
         DD, FF, FF, 1, 1, 0,0,0,0,0,0, 0, 1 /*gelu*/, 1.f);
    gemm(hid_bf, dW[b][9], decP[b][15], yf, yf, nullptr, RR, DD, FF,
         FF, DD, DD, 1, 1, 0,0,0,0,0,0, 0, 0, 1.f);
  }
  ln_kernel<<<RR, 256, 0, stream>>>(yf, dec_lng, dec_lnb, nrm_bf, DD);

  float* logits = (float*)d_out;
  gemm(nrm_bf, projw_bf, proj_b, nullptr, logits, nullptr, RR, VV, DD,
       DD, VV, VV, 1, 1, 0,0,0,0,0,0, 0, 0, 1.f);
  logsoftmax_kernel<<<RR, 256, 0, stream>>>(logits, VV);
}
